// MetabolicMultiHeadAttention_14096082665823
// MI455X (gfx1250) — matrix operand order repaired and checked by reading, not compiled, not run on hardware
//
#include <hip/hip_runtime.h>

typedef __bf16 bf16;
typedef __attribute__((ext_vector_type(16))) __bf16 v16bf;
typedef __attribute__((ext_vector_type(8)))  __bf16 v8bf;
typedef __attribute__((ext_vector_type(8)))  float  v8f;
typedef __attribute__((ext_vector_type(4)))  float  v4f;

#define DEV __device__ __forceinline__

static constexpr int S  = 2048;   // sequence
static constexpr int D  = 1024;   // model dim
static constexpr int H  = 16;     // heads
static constexpr int HD = 64;     // head dim
static constexpr int E  = 65536;  // edges
static constexpr int DH = 512;    // MLP hidden
static constexpr int KP = 1056;   // nt_in K padded to multiple of 32 (1026 -> 1056)

// ---------------------------------------------------------------------------
// WMMA fragment helpers (wave32, 16x16x32 bf16, f32 accumulate)
// A 16x32 (MxK): lane L holds row M=L%16; VGPR0-3 = K[(L/16)*8 .. +7],
//                VGPR4-7 = K[16+(L/16)*8 .. +7]  (contiguous 16B each)
// B 32x16 (KxN) sourced from row-major Bt[n][k] (i.e. B^T): lane L holds
//                col N=L%16, K[(L/16)*8 .. +7] and K[16+(L/16)*8 .. +7] (16B each)
// C 16x16: acc[r] -> (M = r + 8*(L/16), N = L%16)
// ---------------------------------------------------------------------------
DEV v16bf load_a_frag(const bf16* A, int lda, int m0, int k0, int lane) {
  int m  = m0 + (lane & 15);
  int kb = k0 + ((lane >> 4) << 3);
  const bf16* p = A + (size_t)m * lda + kb;
  v8bf lo = *(const v8bf*)p;
  v8bf hi = *(const v8bf*)(p + 16);
  v16bf r;
  #pragma unroll
  for (int i = 0; i < 8; ++i) { r[i] = lo[i]; r[8 + i] = hi[i]; }
  return r;
}

DEV v16bf load_b_frag(const bf16* Bt, int ldb, int n0, int k0, int lane) {
  int n  = n0 + (lane & 15);
  int kb = k0 + ((lane >> 4) << 3);
  const bf16* p = Bt + (size_t)n * ldb + kb;
  v8bf lo = *(const v8bf*)p;
  v8bf hi = *(const v8bf*)(p + 16);
  v16bf r;
  #pragma unroll
  for (int i = 0; i < 8; ++i) { r[i] = lo[i]; r[8 + i] = hi[i]; }
  return r;
}

DEV v8f wmma_bf16(v16bf a, v16bf b, v8f c) {
  return __builtin_amdgcn_wmma_f32_16x16x32_bf16(false, a, false, b, (short)0, c,
                                                 false, false);
}

// ---------------------------------------------------------------------------
// Conversion / packing kernels
// ---------------------------------------------------------------------------
__global__ void conv_bf16_kernel(const float* __restrict__ src,
                                 bf16* __restrict__ dst, int n) {
  int i = blockIdx.x * blockDim.x + threadIdx.x;
  if (i < n) dst[i] = (bf16)src[i];
}

__global__ void build_ntin_kernel(const float* __restrict__ x,
                                  const float* __restrict__ nt,
                                  bf16* __restrict__ ntin) {
  int i = blockIdx.x * blockDim.x + threadIdx.x;
  if (i >= S * KP) return;
  int r = i / KP, c = i % KP;
  float v = (c < D) ? x[(size_t)r * D + c]
          : (c < D + 2) ? nt[(size_t)r * 2 + (c - D)] : 0.f;
  ntin[i] = (bf16)v;
}

__global__ void build_wn1p_kernel(const float* __restrict__ w,
                                  bf16* __restrict__ wp) {
  int i = blockIdx.x * blockDim.x + threadIdx.x;
  if (i >= DH * KP) return;
  int r = i / KP, c = i % KP;
  wp[i] = (bf16)((c < D + 2) ? w[(size_t)r * (D + 2) + c] : 0.f);
}

// V[s][d] (bf16, row-major) -> Vt[d][s]
__global__ void transpose_bf16_kernel(const bf16* __restrict__ V,
                                      bf16* __restrict__ Vt) {
  __shared__ bf16 t[32][33];
  int c0 = blockIdx.x * 32;  // d
  int r0 = blockIdx.y * 32;  // s
  int tx = threadIdx.x & 31, ty = threadIdx.x >> 5;  // 32x8
  #pragma unroll
  for (int it = 0; it < 4; ++it)
    t[ty + it * 8][tx] = V[(size_t)(r0 + ty + it * 8) * D + c0 + tx];
  __syncthreads();
  #pragma unroll
  for (int it = 0; it < 4; ++it)
    Vt[(size_t)(c0 + ty + it * 8) * S + r0 + tx] = t[tx][ty + it * 8];
}

// ---------------------------------------------------------------------------
// Generic GEMM: C_bf16[m][n] = act( A[M,K] @ Bt[N,K]^T + bias[n] )
// block = 128 threads = 4 waves; wave -> 16x64 strip; grid (N/64, M/64)
// ---------------------------------------------------------------------------
__global__ __launch_bounds__(128)
void gemm_bias_act_kernel(const bf16* __restrict__ A, int lda,
                          const bf16* __restrict__ Bt, int ldb,
                          const float* __restrict__ bias,
                          bf16* __restrict__ C, int ldc, int K, int relu) {
  int lane = threadIdx.x & 31, wave = threadIdx.x >> 5;
  int m0 = (blockIdx.y * 4 + wave) * 16;
  int n0 = blockIdx.x * 64;
  v8f acc[4] = {};
  for (int k0 = 0; k0 < K; k0 += 32) {
    v16bf a = load_a_frag(A, lda, m0, k0, lane);
    #pragma unroll
    for (int j = 0; j < 4; ++j) {
      v16bf b = load_b_frag(Bt, ldb, n0 + j * 16, k0, lane);
      acc[j] = wmma_bf16(a, b, acc[j]);
    }
  }
  #pragma unroll
  for (int j = 0; j < 4; ++j) {
    int n = n0 + j * 16 + (lane & 15);
    float bv = bias[n];
    #pragma unroll
    for (int r = 0; r < 8; ++r) {
      int m = m0 + r + ((lane >> 4) << 3);
      float v = acc[j][r] + bv;
      if (relu) v = v > 0.f ? v : 0.f;
      C[(size_t)m * ldc + n] = (bf16)v;
    }
  }
}

// Small GEMM, N == 16 (MLP head logits), f32 output, ldc = 16
__global__ __launch_bounds__(128)
void gemm_n16_kernel(const bf16* __restrict__ A, int lda,
                     const bf16* __restrict__ Bt, int ldb,
                     const float* __restrict__ bias,
                     float* __restrict__ C, int K) {
  int lane = threadIdx.x & 31, wave = threadIdx.x >> 5;
  int m0 = (blockIdx.x * 4 + wave) * 16;
  v8f acc = {};
  for (int k0 = 0; k0 < K; k0 += 32) {
    v16bf a = load_a_frag(A, lda, m0, k0, lane);
    v16bf b = load_b_frag(Bt, ldb, 0, k0, lane);
    acc = wmma_bf16(a, b, acc);
  }
  int n = lane & 15;
  float bv = bias[n];
  #pragma unroll
  for (int r = 0; r < 8; ++r) {
    int m = m0 + r + ((lane >> 4) << 3);
    C[(size_t)m * 16 + n] = acc[r] + bv;
  }
}

// softmax over the 16 heads, per node
__global__ void softmax16_kernel(const float* __restrict__ logits,
                                 float* __restrict__ ntw) {
  int i = blockIdx.x * blockDim.x + threadIdx.x;
  if (i >= S) return;
  const float* p = logits + (size_t)i * 16;
  float mx = p[0];
  #pragma unroll
  for (int h = 1; h < 16; ++h) mx = fmaxf(mx, p[h]);
  float e[16], sm = 0.f;
  #pragma unroll
  for (int h = 0; h < 16; ++h) { e[h] = __expf(p[h] - mx); sm += e[h]; }
  float inv = 1.f / sm;
  #pragma unroll
  for (int h = 0; h < 16; ++h) ntw[(size_t)i * 16 + h] = e[h] * inv;
}

// ---------------------------------------------------------------------------
// Stoichiometric mask
// ---------------------------------------------------------------------------
__global__ void zero_f32_kernel(float* __restrict__ p, size_t n) {
  size_t i = (size_t)blockIdx.x * blockDim.x + threadIdx.x;
  size_t stride = (size_t)gridDim.x * blockDim.x;
  for (; i < n; i += stride) p[i] = 0.f;
}

__global__ void scatter_edges_kernel(const int* __restrict__ ei,
                                     const float* __restrict__ st,
                                     float* __restrict__ mask) {
  int i = blockIdx.x * blockDim.x + threadIdx.x;
  if (i < E) mask[(size_t)ei[i] * S + ei[E + i]] = st[i];
}

// ---------------------------------------------------------------------------
// Scores: raw[h][q][k] = (Q_h @ K_h^T)/8 * ntw[k][h] * mask[q][k]
// grid (S/64 in k, S/64 in q, H); wave -> 16(q) x 64(k)
// ---------------------------------------------------------------------------
__global__ __launch_bounds__(128)
void scores_kernel(const bf16* __restrict__ Q, const bf16* __restrict__ Km,
                   const float* __restrict__ ntw, const float* __restrict__ mask,
                   float* __restrict__ attn) {
  int lane = threadIdx.x & 31, wave = threadIdx.x >> 5;
  int h  = blockIdx.z;
  int m0 = (blockIdx.y * 4 + wave) * 16;
  int n0 = blockIdx.x * 64;
  const bf16* A  = Q  + h * HD;
  const bf16* Bt = Km + h * HD;
  v8f acc[4] = {};
  #pragma unroll
  for (int k0 = 0; k0 < HD; k0 += 32) {
    v16bf a = load_a_frag(A, D, m0, k0, lane);
    #pragma unroll
    for (int j = 0; j < 4; ++j) {
      v16bf b = load_b_frag(Bt, D, n0 + j * 16, k0, lane);
      acc[j] = wmma_bf16(a, b, acc[j]);
    }
  }
  #pragma unroll
  for (int j = 0; j < 4; ++j) {
    int n = n0 + j * 16 + (lane & 15);
    float w = 0.125f * ntw[(size_t)n * H + h];   // 1/sqrt(64) fused
    #pragma unroll
    for (int r = 0; r < 8; ++r) {
      int m = m0 + r + ((lane >> 4) << 3);
      float v = acc[j][r] * w * mask[(size_t)m * S + n];
      attn[((size_t)h * S + m) * S + n] = v;
    }
  }
}

// Per-row (h*S+q) max and sum(exp(v-max)) over the 2048 keys
__global__ __launch_bounds__(256)
void rowstats_kernel(const float* __restrict__ attn,
                     float* __restrict__ rmax, float* __restrict__ rsum) {
  __shared__ float red[256];
  size_t row = blockIdx.x;
  const float* p = attn + row * (size_t)S;
  float mx = -1e30f;
  for (int i = threadIdx.x; i < S; i += 256) mx = fmaxf(mx, p[i]);
  red[threadIdx.x] = mx; __syncthreads();
  for (int s = 128; s > 0; s >>= 1) {
    if (threadIdx.x < s) red[threadIdx.x] = fmaxf(red[threadIdx.x], red[threadIdx.x + s]);
    __syncthreads();
  }
  float mxv = red[0]; __syncthreads();
  float sm = 0.f;
  for (int i = threadIdx.x; i < S; i += 256) sm += __expf(p[i] - mxv);
  red[threadIdx.x] = sm; __syncthreads();
  for (int s = 128; s > 0; s >>= 1) {
    if (threadIdx.x < s) red[threadIdx.x] += red[threadIdx.x + s];
    __syncthreads();
  }
  if (threadIdx.x == 0) { rmax[row] = mxv; rsum[row] = red[0]; }
}

// attn = softmax(raw) written back in-place (each element exactly once),
// plus out_h = attn @ V_h via WMMA. grid (H, S/64); wave -> 16 q-rows, N=64.
__global__ __launch_bounds__(128)
void attnv_kernel(float* __restrict__ attn,
                  const float* __restrict__ rmax, const float* __restrict__ rsum,
                  const bf16* __restrict__ Vt, bf16* __restrict__ attnout) {
  int lane = threadIdx.x & 31, wave = threadIdx.x >> 5;
  int h  = blockIdx.x;
  int q0 = (blockIdx.y * 4 + wave) * 16;
  int mrow = q0 + (lane & 15);
  size_t rowix = (size_t)h * S + mrow;
  float mx  = rmax[rowix];
  float inv = 1.f / rsum[rowix];
  float* prow = attn + rowix * (size_t)S;
  v8f acc[4] = {};
  for (int k0 = 0; k0 < S; k0 += 32) {
    int kb = k0 + ((lane >> 4) << 3);
    float* p0 = prow + kb;        // K = kb..kb+7
    float* p1 = prow + kb + 16;   // K = kb+16..kb+23
    v4f f0 = *(v4f*)p0, f1 = *(v4f*)(p0 + 4);
    v4f f2 = *(v4f*)p1, f3 = *(v4f*)(p1 + 4);
    v16bf a;
    #pragma unroll
    for (int i = 0; i < 4; ++i) {
      f0[i] = __expf(f0[i] - mx) * inv;
      f1[i] = __expf(f1[i] - mx) * inv;
      f2[i] = __expf(f2[i] - mx) * inv;
      f3[i] = __expf(f3[i] - mx) * inv;
      a[i] = (bf16)f0[i]; a[4 + i] = (bf16)f1[i];
      a[8 + i] = (bf16)f2[i]; a[12 + i] = (bf16)f3[i];
    }
    *(v4f*)p0 = f0; *(v4f*)(p0 + 4) = f1;   // write normalized attn back
    *(v4f*)p1 = f2; *(v4f*)(p1 + 4) = f3;
    #pragma unroll
    for (int j = 0; j < 4; ++j) {
      v16bf b = load_b_frag(Vt, S, h * HD + j * 16, k0, lane);
      acc[j] = wmma_bf16(a, b, acc[j]);
    }
  }
  #pragma unroll
  for (int j = 0; j < 4; ++j) {
    int n = h * HD + j * 16 + (lane & 15);
    #pragma unroll
    for (int r = 0; r < 8; ++r) {
      int m = q0 + r + ((lane >> 4) << 3);
      attnout[(size_t)m * D + n] = (bf16)acc[j][r];
    }
  }
}

// preln[m][n] = attnout @ Wo^T + bo[n] + x[m][n]
__global__ __launch_bounds__(128)
void outproj_kernel(const bf16* __restrict__ A, const bf16* __restrict__ Wot,
                    const float* __restrict__ bo, const float* __restrict__ x,
                    float* __restrict__ preln) {
  int lane = threadIdx.x & 31, wave = threadIdx.x >> 5;
  int m0 = (blockIdx.y * 4 + wave) * 16;
  int n0 = blockIdx.x * 64;
  v8f acc[4] = {};
  for (int k0 = 0; k0 < D; k0 += 32) {
    v16bf a = load_a_frag(A, D, m0, k0, lane);
    #pragma unroll
    for (int j = 0; j < 4; ++j) {
      v16bf b = load_b_frag(Wot, D, n0 + j * 16, k0, lane);
      acc[j] = wmma_bf16(a, b, acc[j]);
    }
  }
  #pragma unroll
  for (int j = 0; j < 4; ++j) {
    int n = n0 + j * 16 + (lane & 15);
    float bv = bo[n];
    #pragma unroll
    for (int r = 0; r < 8; ++r) {
      int m = m0 + r + ((lane >> 4) << 3);
      preln[(size_t)m * D + n] = acc[j][r] + bv + x[(size_t)m * D + n];
    }
  }
}

__global__ __launch_bounds__(256)
void layernorm_kernel(const float* __restrict__ pre, const float* __restrict__ g,
                      const float* __restrict__ b, float* __restrict__ y) {
  __shared__ float red[256];
  const float* p = pre + (size_t)blockIdx.x * D;
  float s = 0.f;
  for (int i = threadIdx.x; i < D; i += 256) s += p[i];
  red[threadIdx.x] = s; __syncthreads();
  for (int st = 128; st > 0; st >>= 1) {
    if (threadIdx.x < st) red[threadIdx.x] += red[threadIdx.x + st];
    __syncthreads();
  }
  float mean = red[0] * (1.f / D); __syncthreads();
  float v = 0.f;
  for (int i = threadIdx.x; i < D; i += 256) { float d0 = p[i] - mean; v += d0 * d0; }
  red[threadIdx.x] = v; __syncthreads();
  for (int st = 128; st > 0; st >>= 1) {
    if (threadIdx.x < st) red[threadIdx.x] += red[threadIdx.x + st];
    __syncthreads();
  }
  float rstd = rsqrtf(red[0] * (1.f / D) + 1e-5f);
  for (int i = threadIdx.x; i < D; i += 256)
    y[(size_t)blockIdx.x * D + i] = (p[i] - mean) * rstd * g[i] + b[i];
}

// ---------------------------------------------------------------------------
extern "C" void kernel_launch(void* const* d_in, const int* in_sizes, int n_in,
                              void* d_out, int out_size, void* d_ws, size_t ws_size,
                              hipStream_t stream) {
  const float* x      = (const float*)d_in[0];
  const float* ntypes = (const float*)d_in[1];
  const float* stoich = (const float*)d_in[2];
  const float* Wq = (const float*)d_in[3];  const float* bq = (const float*)d_in[4];
  const float* Wk = (const float*)d_in[5];  const float* bk = (const float*)d_in[6];
  const float* Wv = (const float*)d_in[7];  const float* bv = (const float*)d_in[8];
  const float* Wo = (const float*)d_in[9];  const float* bo = (const float*)d_in[10];
  const float* Wn1 = (const float*)d_in[11]; const float* bn1 = (const float*)d_in[12];
  const float* Wn2 = (const float*)d_in[13]; const float* bn2 = (const float*)d_in[14];
  const float* ln_g = (const float*)d_in[15]; const float* ln_b = (const float*)d_in[16];
  const int*   edges = (const int*)d_in[17];   // [2, E] (JAX default int32)

  float* y    = (float*)d_out;                 // [S, D]
  float* attn = y + (size_t)S * D;             // [H, S, S]

  // workspace carve-up (~64 MiB total)
  char* wsp = (char*)d_ws;
  auto alloc = [&](size_t bytes) -> void* {
    void* r = (void*)wsp;
    wsp += (bytes + 255) & ~(size_t)255;
    return r;
  };
  bf16* xb    = (bf16*)alloc((size_t)S * D * 2);
  bf16* Wqb   = (bf16*)alloc((size_t)D * D * 2);
  bf16* Wkb   = (bf16*)alloc((size_t)D * D * 2);
  bf16* Wvb   = (bf16*)alloc((size_t)D * D * 2);
  bf16* Wob   = (bf16*)alloc((size_t)D * D * 2);
  bf16* ntin  = (bf16*)alloc((size_t)S * KP * 2);
  bf16* Wn1b  = (bf16*)alloc((size_t)DH * KP * 2);
  bf16* Wn2b  = (bf16*)alloc((size_t)H * DH * 2);
  bf16* Qb    = (bf16*)alloc((size_t)S * D * 2);
  bf16* Kb    = (bf16*)alloc((size_t)S * D * 2);
  bf16* Vb    = (bf16*)alloc((size_t)S * D * 2);
  bf16* Vtb   = (bf16*)alloc((size_t)D * S * 2);
  bf16* hid   = (bf16*)alloc((size_t)S * DH * 2);
  float* logits = (float*)alloc((size_t)S * H * 4);
  float* ntw    = (float*)alloc((size_t)S * H * 4);
  float* mask   = (float*)alloc((size_t)S * S * 4);
  float* rmax   = (float*)alloc((size_t)H * S * 4);
  float* rsum   = (float*)alloc((size_t)H * S * 4);
  bf16*  attnout = (bf16*)alloc((size_t)S * D * 2);
  float* preln   = (float*)alloc((size_t)S * D * 4);
  (void)in_sizes; (void)n_in; (void)out_size; (void)ws_size;

  const int T = 256;
  // 1) bf16 packing
  conv_bf16_kernel<<<(S * D + T - 1) / T, T, 0, stream>>>(x, xb, S * D);
  conv_bf16_kernel<<<(D * D + T - 1) / T, T, 0, stream>>>(Wq, Wqb, D * D);
  conv_bf16_kernel<<<(D * D + T - 1) / T, T, 0, stream>>>(Wk, Wkb, D * D);
  conv_bf16_kernel<<<(D * D + T - 1) / T, T, 0, stream>>>(Wv, Wvb, D * D);
  conv_bf16_kernel<<<(D * D + T - 1) / T, T, 0, stream>>>(Wo, Wob, D * D);
  conv_bf16_kernel<<<(H * DH + T - 1) / T, T, 0, stream>>>(Wn2, Wn2b, H * DH);
  build_ntin_kernel<<<(S * KP + T - 1) / T, T, 0, stream>>>(x, ntypes, ntin);
  build_wn1p_kernel<<<(DH * KP + T - 1) / T, T, 0, stream>>>(Wn1, Wn1b);

  // 2) QKV projections (bf16 WMMA, f32 acc)
  gemm_bias_act_kernel<<<dim3(D / 64, S / 64), 128, 0, stream>>>(xb, D, Wqb, D, bq, Qb, D, D, 0);
  gemm_bias_act_kernel<<<dim3(D / 64, S / 64), 128, 0, stream>>>(xb, D, Wkb, D, bk, Kb, D, D, 0);
  gemm_bias_act_kernel<<<dim3(D / 64, S / 64), 128, 0, stream>>>(xb, D, Wvb, D, bv, Vb, D, D, 0);
  transpose_bf16_kernel<<<dim3(D / 32, S / 32), 256, 0, stream>>>(Vb, Vtb);

  // 3) node-type MLP + head softmax
  gemm_bias_act_kernel<<<dim3(DH / 64, S / 64), 128, 0, stream>>>(ntin, KP, Wn1b, KP, bn1, hid, DH, KP, 1);
  gemm_n16_kernel<<<S / 64, 128, 0, stream>>>(hid, DH, Wn2b, DH, bn2, logits, DH);
  softmax16_kernel<<<(S + T - 1) / T, T, 0, stream>>>(logits, ntw);

  // 4) stoichiometric mask
  zero_f32_kernel<<<2048, T, 0, stream>>>(mask, (size_t)S * S);
  scatter_edges_kernel<<<E / T, T, 0, stream>>>(edges, stoich, mask);

  // 5) masked scores -> raw attn (in d_out region)
  scores_kernel<<<dim3(S / 64, S / 64, H), 128, 0, stream>>>(Qb, Kb, ntw, mask, attn);

  // 6) softmax stats, then fused normalize + attn@V
  rowstats_kernel<<<H * S, 256, 0, stream>>>(attn, rmax, rsum);
  attnv_kernel<<<dim3(H, S / 64), 128, 0, stream>>>(attn, rmax, rsum, Vtb, attnout);

  // 7) output projection + residual, then LayerNorm -> y
  outproj_kernel<<<dim3(D / 64, S / 64), 128, 0, stream>>>(attnout, Wob, bo, x, preln);
  layernorm_kernel<<<S, 256, 0, stream>>>(preln, ln_g, ln_b, y);
}
